// OAIAttention_21354577395899
// MI455X (gfx1250) — matrix operand order repaired and checked by reading, not compiled, not run on hardware
//
#include <hip/hip_runtime.h>
#include <hip/hip_bf16.h>

// ---------------- problem constants ----------------
constexpr int T_    = 4096;
constexpr int HID_  = 2880;
constexpr int HQ_   = 64;
constexpr int HKV_  = 8;
constexpr int D_    = 64;
constexpr int W_    = 128;
constexpr int NB_   = T_ / W_;                 // 32
constexpr int QKV_OUT = (HQ_ + 2 * HKV_) * D_; // 5120
constexpr float SCALE_ = 0.125f;               // D^-0.5
constexpr float NEG_   = -1e30f;

// ---------------- vector types ----------------
typedef __attribute__((ext_vector_type(16))) __bf16 v16bf;
typedef __attribute__((ext_vector_type(8)))  __bf16 v8bf;
typedef __attribute__((ext_vector_type(8)))  float  v8f;
typedef __attribute__((ext_vector_type(4)))  float  v4f;
typedef __attribute__((ext_vector_type(4)))  unsigned int v4u;
typedef __attribute__((ext_vector_type(8)))  int    v8i;
typedef __attribute__((ext_vector_type(4)))  int    v4i;

union V16 { v16bf v; v8bf h[2]; };

static __device__ __forceinline__ v8f wmma_bf16(const V16& a, const V16& b, v8f c) {
  return __builtin_amdgcn_wmma_f32_16x16x32_bf16(
      /*neg_a=*/false, a.v, /*neg_b=*/false, b.v,
      /*c_mod=*/(short)0, c, /*reuse_a=*/false, /*reuse_b=*/false);
}

// CDNA5 async LDS copy: 16B per lane, global -> LDS, tracked by ASYNCcnt.
static __device__ __forceinline__ void async_copy_b128(void* lds_ptr, const void* gptr) {
  unsigned loff = (unsigned)(unsigned long long)lds_ptr;  // low 32 bits = LDS offset
  unsigned long long ga = (unsigned long long)gptr;
  asm volatile("global_load_async_to_lds_b128 %0, %1, off"
               :: "v"(loff), "v"(ga) : "memory");
}
static __device__ __forceinline__ void wait_async0() {
  asm volatile("s_wait_asynccnt 0x0" ::: "memory");
}

// CDNA5 Tensor Data Mover: wave-level 2D tile DMA global -> LDS (TENSORcnt).
// width/rows in 2-byte elements; stride = row stride in elements.
// pad=true inserts 16B of LDS padding after each 64B row (-> 40-elem rows).
static __device__ __forceinline__ void tdm_load_2d(
    unsigned lds_off, const void* gaddr, unsigned width, unsigned rows,
    unsigned stride, bool pad) {
  unsigned long long ga = (unsigned long long)gaddr;
  // D# group0: count=1 | lds_addr | global_addr[56:0] | type=2 (image)
  v4u g0 = { 1u, lds_off, (unsigned)ga, (unsigned)(ga >> 32) | (2u << 30) };
  // D# group1: data_size=1(2B) [+ pad_enable, pad_interval=16dw, pad_amount=4dw]
  unsigned d0 = (1u << 16) | (pad ? ((1u << 20) | (3u << 22) | (3u << 25)) : 0u);
  v8i g1 = { (int)d0,
             (int)(width << 16),                        // tensor_dim0[15:0]
             (int)((width >> 16) | (rows << 16)),       // tensor_dim0 hi | tensor_dim1 lo
             (int)((rows >> 16) | (width << 16)),       // tensor_dim1 hi | tile_dim0
             (int)(rows & 0xffffu),                     // tile_dim1 (tile_dim2=0)
             (int)stride,                               // tensor_dim0_stride[31:0]
             0, 0 };
  v4i gz = { 0, 0, 0, 0 };
#if defined(__clang_major__) && __clang_major__ >= 23
  v8i gz8 = { 0, 0, 0, 0, 0, 0, 0, 0 };
  __builtin_amdgcn_tensor_load_to_lds(g0, g1, gz, gz, gz8, 0);
#else
  __builtin_amdgcn_tensor_load_to_lds(g0, g1, gz, gz, 0);
#endif
}
static __device__ __forceinline__ void wait_tensor0() {
  __builtin_amdgcn_s_wait_tensorcnt(0);
}

// ---------------- helper kernels ----------------
__global__ __launch_bounds__(256) void yarn_table_kernel(float* __restrict__ cosT,
                                                         float* __restrict__ sinT) {
  int idx = blockIdx.x * 256 + threadIdx.x;
  if (idx >= T_ * 32) return;
  int t = idx >> 5, i = idx & 31;
  float freq      = __powf(150000.f, (float)i / 32.f);
  float inv_extra = 1.f / freq;
  float inv_inter = 1.f / (32.f * freq);
  const float PI2 = 6.28318530717958647692f;
  float logb = logf(150000.f);
  float lowf  = floorf(64.f * logf(4096.f / (32.f * PI2)) / (2.f * logb));
  float highf = ceilf (64.f * logf(4096.f / ( 1.f * PI2)) / (2.f * logb));
  if (lowf < 0.f) lowf = 0.f;
  if (highf > 63.f) highf = 63.f;
  float ramp = ((float)i - lowf) / fmaxf(highf - lowf, 0.001f);
  ramp = fminf(fmaxf(ramp, 0.f), 1.f);
  float invf   = inv_inter * ramp + inv_extra * (1.f - ramp);
  float mscale = 0.1f * logf(32.f) + 1.f;
  float ang = (float)t * invf;
  cosT[idx] = cosf(ang) * mscale;
  sinT[idx] = sinf(ang) * mscale;
}

__global__ __launch_bounds__(256) void f32_to_bf16_kernel(const float* __restrict__ src,
                                                          __bf16* __restrict__ dst, int n) {
  int idx = blockIdx.x * 256 + threadIdx.x;
  if (idx < n) dst[idx] = (__bf16)src[idx];
}

// f32 [K][N] -> bf16 [N][K] tiled transpose (weights become TDM/async friendly).
__global__ __launch_bounds__(256) void transpose_cvt_kernel(
    const float* __restrict__ src, __bf16* __restrict__ dst, int K, int N) {
  __shared__ float tile[32][33];
  int k0 = blockIdx.y * 32, n0 = blockIdx.x * 32;
  int tx = threadIdx.x & 31, ty = threadIdx.x >> 5;  // 32 x 8
#pragma unroll
  for (int j = 0; j < 32; j += 8)
    tile[ty + j][tx] = src[(size_t)(k0 + ty + j) * N + n0 + tx];
  __syncthreads();
#pragma unroll
  for (int j = 0; j < 32; j += 8)
    dst[(size_t)(n0 + ty + j) * K + k0 + tx] = (__bf16)tile[tx][ty + j];
}

// Apply RoPE to q/k slices of raw qkv and scatter into q_ws/k_ws/v_ws.
__global__ __launch_bounds__(256) void rope_scatter_kernel(
    const __bf16* __restrict__ qkv, const float* __restrict__ cosT,
    const float* __restrict__ sinT, __bf16* __restrict__ qo,
    __bf16* __restrict__ ko, __bf16* __restrict__ vo) {
  const int RQ = T_ * HQ_ * 32;
  const int RK = T_ * HKV_ * 32;
  const int RV = T_ * HKV_ * 64;
  int idx = blockIdx.x * 256 + threadIdx.x;
  if (idx < RQ) {
    int t = idx / (HQ_ * 32);
    int r = idx - t * (HQ_ * 32);
    int hq = r >> 5, i = r & 31;
    const __bf16* row = qkv + (size_t)t * QKV_OUT + hq * 64;
    float x1 = (float)row[i], x2 = (float)row[i + 32];
    float c = cosT[t * 32 + i], s = sinT[t * 32 + i];
    __bf16* dst = qo + ((size_t)t * HQ_ + hq) * 64;
    dst[i]      = (__bf16)(x1 * c - x2 * s);
    dst[i + 32] = (__bf16)(x2 * c + x1 * s);
  } else if (idx < RQ + RK) {
    int j = idx - RQ;
    int t = j / (HKV_ * 32);
    int r = j - t * (HKV_ * 32);
    int hk = r >> 5, i = r & 31;
    const __bf16* row = qkv + (size_t)t * QKV_OUT + HQ_ * 64 + hk * 64;
    float x1 = (float)row[i], x2 = (float)row[i + 32];
    float c = cosT[t * 32 + i], s = sinT[t * 32 + i];
    __bf16* dst = ko + ((size_t)t * HKV_ + hk) * 64;
    dst[i]      = (__bf16)(x1 * c - x2 * s);
    dst[i + 32] = (__bf16)(x2 * c + x1 * s);
  } else if (idx < RQ + RK + RV) {
    int j = idx - RQ - RK;
    int t = j / (HKV_ * 64);
    int r = j - t * (HKV_ * 64);
    vo[(size_t)t * HKV_ * 64 + r] =
        qkv[(size_t)t * QKV_OUT + (HQ_ + HKV_) * 64 + r];
  }
}

// ---------------- bf16 WMMA GEMM ----------------
// C[M,N] = A[M,K](bf16 row-major) * BT[N,K](bf16, pre-transposed) + bias
// Double-buffered LDS; tiles staged by the Tensor Data Mover (wave 0 issues,
// everyone synchronizes via s_wait_tensorcnt + workgroup barrier).
#define BM 256
#define BN 64
#define BK 32
#define ASTR 40  // bf16 elems per LDS row: 32 data + 8 pad (matches TDM pad fields)
#define BSTR 40

__global__ __launch_bounds__(256) void gemm_bf16_kernel(
    const __bf16* __restrict__ A, const __bf16* __restrict__ BT,
    const float* __restrict__ bias, float* __restrict__ Cf,
    __bf16* __restrict__ Cb, int M, int N, int K, int store_f32) {
  __shared__ __bf16 aLds[2][BM * ASTR];
  __shared__ __bf16 bLds[2][BN * BSTR];

  const int tid  = threadIdx.x;
  const int lane = tid & 31;
  const int wave = tid >> 5;
  const int h    = (lane >> 4) & 1;
  const int ln   = lane & 15;
  const int wm   = (wave >> 1) * 64;  // 4 waves along M
  const int wn   = (wave & 1) * 32;   // 2 waves along N
  const int bm   = blockIdx.y * BM;
  const int bn   = blockIdx.x * BN;

  const __bf16* aBase = A + (size_t)bm * K;
  const __bf16* bBase = BT + (size_t)bn * K;
  const int NK = K / BK;

  auto stage = [&](int kt, int buf) {
    if (wave == 0) {
      tdm_load_2d((unsigned)(unsigned long long)&aLds[buf][0],
                  aBase + kt * BK, BK, BM, (unsigned)K, true);
      tdm_load_2d((unsigned)(unsigned long long)&bLds[buf][0],
                  bBase + kt * BK, BK, BN, (unsigned)K, true);
    }
  };

  v8f acc[4][2] = {};
  stage(0, 0);
  for (int kt = 0; kt < NK; ++kt) {
    wait_tensor0();     // wave0: stage kt landed (others: no-op)
    __syncthreads();    // publish LDS to all waves
    if (kt + 1 < NK) stage(kt + 1, (kt + 1) & 1);  // overlap DMA with compute

    const __bf16* aB = aLds[kt & 1];
    const __bf16* bB = bLds[kt & 1];
    V16 afr[4], bfr[2];
#pragma unroll
    for (int mt = 0; mt < 4; ++mt) {
      int r = wm + mt * 16 + ln;
      afr[mt].h[0] = *(const v8bf*)&aB[r * ASTR + 8 * h];
      afr[mt].h[1] = *(const v8bf*)&aB[r * ASTR + 16 + 8 * h];
    }
#pragma unroll
    for (int nt = 0; nt < 2; ++nt) {
      int n = wn + nt * 16 + ln;
      bfr[nt].h[0] = *(const v8bf*)&bB[n * BSTR + 8 * h];
      bfr[nt].h[1] = *(const v8bf*)&bB[n * BSTR + 16 + 8 * h];
    }
#pragma unroll
    for (int mt = 0; mt < 4; ++mt)
#pragma unroll
      for (int nt = 0; nt < 2; ++nt)
        acc[mt][nt] = wmma_bf16(afr[mt], bfr[nt], acc[mt][nt]);
  }

  // epilogue: + bias, store f32 or bf16 (C layout: M = v + 8h, N = ln)
#pragma unroll
  for (int mt = 0; mt < 4; ++mt)
#pragma unroll
    for (int nt = 0; nt < 2; ++nt) {
      int col = bn + wn + nt * 16 + ln;
      float bv = bias[col];
#pragma unroll
      for (int v = 0; v < 8; ++v) {
        int row = bm + wm + mt * 16 + v + 8 * h;
        float val = acc[mt][nt][v] + bv;
        if (store_f32) Cf[(size_t)row * N + col] = val;
        else           Cb[(size_t)row * N + col] = (__bf16)val;
      }
    }
}

// ---------------- windowed attention with sinks ----------------
// one block per (time-block b, kv-head); 8 waves, one q-head per wave.
constexpr int VSTR = 264;  // padded key stride for transposed V in LDS

__global__ __launch_bounds__(256) void attn_kernel(
    const __bf16* __restrict__ q, const __bf16* __restrict__ k,
    const __bf16* __restrict__ v, const float* __restrict__ sinks,
    __bf16* __restrict__ o) {
  __shared__ __bf16 kLds[256 * 64];        // [key][d]
  __shared__ __bf16 vLds[64 * VSTR];       // [d][key] (transposed)
  __shared__ __bf16 pLds[8][16 * 32];      // per-wave P re-layout scratch

  const int b    = blockIdx.x >> 3;
  const int kvh  = blockIdx.x & 7;
  const int tid  = threadIdx.x;
  const int lane = tid & 31;
  const int wave = tid >> 5;
  const int h    = (lane >> 4) & 1;
  const int ln   = lane & 15;
  const int head = kvh * 8 + wave;

  // ---- stage K (async direct-to-LDS) and V (VGPR transpose); one key per thread
  {
    int key  = tid;                       // 0..255
    int kpos = (b - 1) * W_ + key;        // absolute key position
    if (kpos >= 0) {
      const __bf16* ks = k + ((size_t)kpos * HKV_ + kvh) * D_;
      const __bf16* vs = v + ((size_t)kpos * HKV_ + kvh) * D_;
#pragma unroll
      for (int d0 = 0; d0 < 64; d0 += 8)
        async_copy_b128(&kLds[key * 64 + d0], ks + d0);
#pragma unroll
      for (int d0 = 0; d0 < 64; d0 += 8) {
        v8bf vv = *(const v8bf*)(vs + d0);
#pragma unroll
        for (int j = 0; j < 8; ++j) vLds[(d0 + j) * VSTR + key] = vv[j];
      }
    } else {
      v8bf z = {};
#pragma unroll
      for (int d0 = 0; d0 < 64; d0 += 8) {
        *(v8bf*)&kLds[key * 64 + d0] = z;
#pragma unroll
        for (int j = 0; j < 8; ++j) vLds[(d0 + j) * VSTR + key] = (__bf16)0.f;
      }
    }
  }
  wait_async0();
  __syncthreads();

  const float sink = sinks[head];

  for (int mblk = 0; mblk < 8; ++mblk) {
    // Q fragments for rows [mblk*16, mblk*16+16): A layout, 2 K-steps of 32
    V16 aq[2];
    {
      int t = b * W_ + mblk * 16 + ln;
      const __bf16* qs = q + ((size_t)t * HQ_ + head) * D_;
#pragma unroll
      for (int ks2 = 0; ks2 < 2; ++ks2) {
        aq[ks2].h[0] = *(const v8bf*)(qs + ks2 * 32 + 8 * h);
        aq[ks2].h[1] = *(const v8bf*)(qs + ks2 * 32 + 16 + 8 * h);
      }
      if (mblk + 1 < 8) __builtin_prefetch(qs + (size_t)16 * HQ_ * D_, 0, 1);
    }

    v8f oacc[4] = {};
    float mr[8], lr[8];
#pragma unroll
    for (int i = 0; i < 8; ++i) { mr[i] = NEG_; lr[i] = 0.f; }

    for (int n2 = 0; n2 < 8; ++n2) {  // 32 keys per step
      int kb = n2 * 32;
      // ---- scores S = Q * K^T for two adjacent 16-key tiles
      v8f s[2] = {};
#pragma unroll
      for (int j = 0; j < 2; ++j) {
        int key = kb + 16 * j + ln;
#pragma unroll
        for (int ks2 = 0; ks2 < 2; ++ks2) {
          V16 bk;
          bk.h[0] = *(const v8bf*)&kLds[key * 64 + ks2 * 32 + 8 * h];
          bk.h[1] = *(const v8bf*)&kLds[key * 64 + ks2 * 32 + 16 + 8 * h];
          s[j] = wmma_bf16(aq[ks2], bk, s[j]);
        }
      }
      // ---- scale + sliding-window mask
#pragma unroll
      for (int j = 0; j < 2; ++j) {
        int ki   = kb + 16 * j + ln;
        int kpos = (b - 1) * W_ + ki;
#pragma unroll
        for (int vi = 0; vi < 8; ++vi) {
          int qi = mblk * 16 + vi + 8 * h;
          bool ok = (ki > qi) && (ki <= qi + W_) && (kpos >= 0);
          s[j][vi] = ok ? s[j][vi] * SCALE_ : NEG_;
        }
      }
      // ---- online softmax update (row reductions across half-wave)
      float p0[8], p1[8], fs[8];
#pragma unroll
      for (int vi = 0; vi < 8; ++vi) {
        float rmax = fmaxf(s[0][vi], s[1][vi]);
#pragma unroll
        for (int off = 1; off < 16; off <<= 1)
          rmax = fmaxf(rmax, __shfl_xor(rmax, off, 32));
        float nm = fmaxf(mr[vi], rmax);
        float f  = __expf(mr[vi] - nm);
        mr[vi] = nm; fs[vi] = f;
        p0[vi] = __expf(s[0][vi] - nm);
        p1[vi] = __expf(s[1][vi] - nm);
        float rs = p0[vi] + p1[vi];
#pragma unroll
        for (int off = 1; off < 16; off <<= 1)
          rs += __shfl_xor(rs, off, 32);
        lr[vi] = lr[vi] * f + rs;
      }
#pragma unroll
      for (int dt = 0; dt < 4; ++dt)
#pragma unroll
        for (int vi = 0; vi < 8; ++vi) oacc[dt][vi] *= fs[vi];

      // ---- P (C layout) -> LDS -> A layout
      __bf16* pw = &pLds[wave][0];
#pragma unroll
      for (int vi = 0; vi < 8; ++vi) {
        int row = vi + 8 * h;
        pw[row * 32 + ln]      = (__bf16)p0[vi];
        pw[row * 32 + 16 + ln] = (__bf16)p1[vi];
      }
      V16 ap;
      ap.h[0] = *(const v8bf*)&pw[ln * 32 + 8 * h];
      ap.h[1] = *(const v8bf*)&pw[ln * 32 + 16 + 8 * h];

      // ---- O += P * V (V staged transposed: contiguous key runs)
#pragma unroll
      for (int dt = 0; dt < 4; ++dt) {
        int dcol = dt * 16 + ln;
        V16 bv;
        bv.h[0] = *(const v8bf*)&vLds[dcol * VSTR + kb + 8 * h];
        bv.h[1] = *(const v8bf*)&vLds[dcol * VSTR + kb + 16 + 8 * h];
        oacc[dt] = wmma_bf16(ap, bv, oacc[dt]);
      }
    }

    // ---- finalize with sink logit, normalize, store
    float g[8];
#pragma unroll
    for (int vi = 0; vi < 8; ++vi) {
      float nm = fmaxf(mr[vi], sink);
      float f  = __expf(mr[vi] - nm);
      float l  = lr[vi] * f + __expf(sink - nm);
      g[vi] = f / l;
    }
#pragma unroll
    for (int dt = 0; dt < 4; ++dt)
#pragma unroll
      for (int vi = 0; vi < 8; ++vi) {
        int t   = b * W_ + mblk * 16 + vi + 8 * h;
        int col = head * 64 + dt * 16 + ln;
        o[(size_t)t * (HQ_ * D_) + col] = (__bf16)(oacc[dt][vi] * g[vi]);
      }
  }
}

// ---------------- host launch ----------------
extern "C" void kernel_launch(void* const* d_in, const int* in_sizes, int n_in,
                              void* d_out, int out_size, void* d_ws, size_t ws_size,
                              hipStream_t stream) {
  (void)in_sizes; (void)n_in; (void)out_size; (void)ws_size;
  const float* hidden = (const float*)d_in[0];
  // d_in[1] = positions (arange(T)) -- implied by index math
  const float* w_qkv = (const float*)d_in[2];
  const float* b_qkv = (const float*)d_in[3];
  const float* w_o   = (const float*)d_in[4];
  const float* b_o   = (const float*)d_in[5];
  const float* sinks = (const float*)d_in[6];
  float* out = (float*)d_out;

  unsigned char* ws = (unsigned char*)d_ws;
  size_t off = 0;
  auto carve = [&](size_t bytes) -> void* {
    void* p = ws + off;
    off += (bytes + 255) & ~(size_t)255;
    return p;
  };
  __bf16* hb      = (__bf16*)carve((size_t)T_ * HID_ * 2);
  __bf16* qkv_raw = (__bf16*)carve((size_t)T_ * QKV_OUT * 2);
  // q-region serves three sequential lives: w_qkv^T -> q(post-RoPE) -> w_o^T
  size_t qreg_bytes = (size_t)T_ * HQ_ * D_ * 2;  // 33.5MB >= both weight^T sizes
  void*   qreg    = carve(qreg_bytes);
  __bf16* k_ws    = (__bf16*)carve((size_t)T_ * HKV_ * D_ * 2);
  __bf16* v_ws    = (__bf16*)carve((size_t)T_ * HKV_ * D_ * 2);
  float*  cosT    = (float*)carve((size_t)T_ * 32 * 4);
  float*  sinT    = (float*)carve((size_t)T_ * 32 * 4);

  __bf16* wqkvT = (__bf16*)qreg;   // life 1
  __bf16* q_ws  = (__bf16*)qreg;   // life 2
  __bf16* woT   = (__bf16*)qreg;   // life 3
  __bf16* o_ws  = qkv_raw;         // alias: qkv_raw dead after rope_scatter

  yarn_table_kernel<<<(T_ * 32 + 255) / 256, 256, 0, stream>>>(cosT, sinT);
  f32_to_bf16_kernel<<<((T_ * HID_) + 255) / 256, 256, 0, stream>>>(hidden, hb, T_ * HID_);

  // w_qkv [HID][QKV_OUT] f32 -> [QKV_OUT][HID] bf16
  transpose_cvt_kernel<<<dim3(QKV_OUT / 32, HID_ / 32), 256, 0, stream>>>(
      w_qkv, wqkvT, HID_, QKV_OUT);

  // QKV projection: [T,HID] x [HID,QKV_OUT] + b_qkv -> bf16 qkv_raw
  gemm_bf16_kernel<<<dim3(QKV_OUT / BN, T_ / BM), 256, 0, stream>>>(
      hb, wqkvT, b_qkv, nullptr, qkv_raw, T_, QKV_OUT, HID_, 0);

  int rope_total = T_ * HQ_ * 32 + T_ * HKV_ * 32 + T_ * HKV_ * 64;
  rope_scatter_kernel<<<(rope_total + 255) / 256, 256, 0, stream>>>(
      qkv_raw, cosT, sinT, q_ws, k_ws, v_ws);

  // attention: one block per (time-block, kv-head)
  attn_kernel<<<NB_ * HKV_, 256, 0, stream>>>(q_ws, k_ws, v_ws, sinks, o_ws);

  // w_o [HQ*D][HID] f32 -> [HID][HQ*D] bf16 (q-region is free after attention)
  transpose_cvt_kernel<<<dim3(HID_ / 32, (HQ_ * D_) / 32), 256, 0, stream>>>(
      w_o, woT, HQ_ * D_, HID_);

  // output projection: [T,HQ*D] x [HQ*D,HID] + b_o -> f32 out
  gemm_bf16_kernel<<<dim3(HID_ / BN, T_ / BM), 256, 0, stream>>>(
      o_ws, woT, b_o, out, nullptr, T_, HID_, HQ_ * D_, 1);
}
